// GroupedQueryAttention_5652176962211
// MI455X (gfx1250) — matrix operand order repaired and checked by reading, not compiled, not run on hardware
//
#include <hip/hip_runtime.h>
#include <hip/hip_bf16.h>
#include <math.h>
#include <stdint.h>

// ---------------------------------------------------------------------------
// Types for CDNA5 WMMA (wave32): v_wmma_f32_16x16x32_bf16
// ---------------------------------------------------------------------------
typedef __attribute__((ext_vector_type(16))) __bf16 v16bf;
typedef __attribute__((ext_vector_type(8)))  __bf16 v8bf;
typedef __attribute__((ext_vector_type(4)))  __bf16 v4bf;
typedef __attribute__((ext_vector_type(8)))  float  v8f;

#define D_MODEL 4096
#define QHEADS  32
#define KVHEADS 8
#define DHEAD   128
#define BATCH   2
#define SEQ     2048
#define KVDIM   (KVHEADS * DHEAD)   // 1024

__device__ __forceinline__ v8f wmma_bf16(v16bf a, v16bf b, v8f c) {
    // 8 args: (neg_a, A, neg_b, B, c_mod, C, reuse_a, reuse_b)
    return __builtin_amdgcn_wmma_f32_16x16x32_bf16(
        false, a, false, b, (short)0, c, false, false);
}

// ---------------------------------------------------------------------------
// CDNA5 async global->LDS copy (ASYNCcnt-tracked, ISA 10.7 / 15.18.3).
// Per-lane: copies 16B from the lane's global address into the lane's LDS
// offset. LDS offset = low 32 bits of the generic pointer (flat->LDS mapping
// truncates to the wave-relative LDS address, ISA 10.2).
// ---------------------------------------------------------------------------
__device__ __forceinline__ void async_copy_b128(void* lds_dst, const void* gsrc) {
    unsigned lds_off = (unsigned)(uintptr_t)lds_dst;
    unsigned long long gaddr = (unsigned long long)(uintptr_t)gsrc;
    asm volatile("global_load_async_to_lds_b128 %0, %1, off"
                 :: "v"(lds_off), "v"(gaddr)
                 : "memory");
}

__device__ __forceinline__ void wait_async_zero() {
    asm volatile("s_wait_asynccnt 0" ::: "memory");
}

// A-matrix 16x32 bf16 fragment (per ISA 7.12.2):
//   lane m = lane&15 ; lanes<16 hold K = {0..7,16..23}, lanes>=16 K = {8..15,24..31}
// `row` points at the start (K=0) of row m.
__device__ __forceinline__ v16bf load_frag_a(const __bf16* row, unsigned lane) {
    const int base = (lane & 16) ? 8 : 0;
    v8bf lo = *(const v8bf*)(row + base);
    v8bf hi = *(const v8bf*)(row + base + 16);
    v16bf r;
#pragma unroll
    for (int i = 0; i < 8; ++i) { r[i] = lo[i]; r[i + 8] = hi[i]; }
    return r;
}

// B-matrix 32x16 bf16 fragment: lane n = lane&15 ; lanes<16 hold K={0..7,16..23},
// lanes>=16 hold K={8..15,24..31} (same order as A fragment).
// `row` points at K=0 of column n's source row (W is [N][K] row-major).
__device__ __forceinline__ v16bf load_frag_b(const __bf16* row, unsigned lane) {
    const int base = (lane & 16) ? 8 : 0;
    v8bf lo = *(const v8bf*)(row + base);
    v8bf hi = *(const v8bf*)(row + base + 16);
    v16bf r;
#pragma unroll
    for (int i = 0; i < 8; ++i) { r[i] = lo[i]; r[i + 8] = hi[i]; }
    return r;
}

// ---------------------------------------------------------------------------
// fp32 -> bf16 conversion (4 elements / thread, 16B loads, 8B stores)
// ---------------------------------------------------------------------------
__global__ __launch_bounds__(256) void f32_to_bf16_kernel(
    const float* __restrict__ in, __bf16* __restrict__ out, long long n) {
    long long i = ((long long)blockIdx.x * blockDim.x + threadIdx.x) * 4;
    if (i + 3 < n) {
        float4 f = *(const float4*)(in + i);
        v4bf b;
        b[0] = (__bf16)f.x; b[1] = (__bf16)f.y;
        b[2] = (__bf16)f.z; b[3] = (__bf16)f.w;
        *(v4bf*)(out + i) = b;
    }
}

// ---------------------------------------------------------------------------
// NT GEMM: C[M,N] = (A[M,K] * W[N,K]^T + bias[N]) * out_scale
// A,W bf16 row-major (K contiguous). 256 threads = 8 waves.
// Block tile 128x128, BK=32, LDS double buffered via ASYNC global->LDS DMA:
// tile k+1 streams into the other buffer while WMMAs consume tile k; the wave
// only pays s_wait_asynccnt 0 + barrier per k-step.
// Wave grid 4(m) x 2(n), wave tile 32x64 => 2x4 WMMA accum tiles.
// ---------------------------------------------------------------------------
#define BM 128
#define BN 128
#define BK 32
#define LDT 40   // padded row stride in bf16 (80B, 16B aligned, avoids conflicts)

template <bool OUT_BF16>
__global__ __launch_bounds__(256) void gemm_bf16_nt(
    const __bf16* __restrict__ A, const __bf16* __restrict__ W,
    const float* __restrict__ bias, void* __restrict__ C,
    int M, int N, int K, float out_scale) {
    __shared__ alignas(16) __bf16 sA[2][BM * LDT];
    __shared__ alignas(16) __bf16 sB[2][BN * LDT];

    const int tid = threadIdx.x;
    const unsigned lane = tid & 31u;
    const int wave = tid >> 5;
    const int wm = wave >> 1;        // 0..3
    const int wn = wave & 1;         // 0..1
    const int bm = blockIdx.y * BM;
    const int bn = blockIdx.x * BN;

    // Cooperative async tile stage: 128 rows x 32 bf16 = 512 16B-chunks, 2/thread.
    auto stage_tile = [&](const __bf16* src, int ld, __bf16* dst) {
#pragma unroll
        for (int i = 0; i < 2; ++i) {
            int c = tid + i * 256;
            int r = c >> 2;
            int s = (c & 3) * 8;
            async_copy_b128(dst + r * LDT + s, src + (size_t)r * ld + s);
        }
    };

    v8f acc[2][4] = {};

    const int nk = K / BK;
    stage_tile(A + (size_t)bm * K, K, sA[0]);
    stage_tile(W + (size_t)bn * K, K, sB[0]);
    wait_async_zero();
    __syncthreads();

    for (int kt = 0; kt < nk; ++kt) {
        const int cur = kt & 1, nxt = cur ^ 1;
        if (kt + 1 < nk) {
            stage_tile(A + (size_t)bm * K + (size_t)(kt + 1) * BK, K, sA[nxt]);
            stage_tile(W + (size_t)bn * K + (size_t)(kt + 1) * BK, K, sB[nxt]);
        }
        v16bf af[2], bfr[4];
#pragma unroll
        for (int mt = 0; mt < 2; ++mt)
            af[mt] = load_frag_a(&sA[cur][(wm * 32 + mt * 16 + (lane & 15)) * LDT], lane);
#pragma unroll
        for (int nt = 0; nt < 4; ++nt)
            bfr[nt] = load_frag_b(&sB[cur][(wn * 64 + nt * 16 + (lane & 15)) * LDT], lane);
#pragma unroll
        for (int mt = 0; mt < 2; ++mt)
#pragma unroll
            for (int nt = 0; nt < 4; ++nt)
                acc[mt][nt] = wmma_bf16(af[mt], bfr[nt], acc[mt][nt]);
        wait_async_zero();   // next-buffer DMA done (overlapped with WMMAs above)
        __syncthreads();
    }

    // Epilogue. C/D layout: vgpr r -> row = r + (lane>=16 ? 8 : 0), col = lane&15.
#pragma unroll
    for (int mt = 0; mt < 2; ++mt) {
#pragma unroll
        for (int nt = 0; nt < 4; ++nt) {
            const int col = bn + wn * 64 + nt * 16 + (int)(lane & 15);
            const float bcol = bias[col];
#pragma unroll
            for (int r = 0; r < 8; ++r) {
                const int row = bm + wm * 32 + mt * 16 + r + ((lane & 16) ? 8 : 0);
                const float v = (acc[mt][nt][r] + bcol) * out_scale;
                if constexpr (OUT_BF16)
                    ((__bf16*)C)[(size_t)row * N + col] = (__bf16)v;
                else
                    ((float*)C)[(size_t)row * N + col] = v;
            }
        }
    }
}

// ---------------------------------------------------------------------------
// Flash attention (causal, GQA). Grid: (S/128, QHEADS, BATCH), 256 thr = 8 waves.
// Wave w owns q rows [qb*128 + w*16, +16). Q pre-scaled by 1/sqrt(dh).
// Per kv-block of 64 keys: K staged [key][d] via async global->LDS DMA
// (QK^T B-frags contiguous); V staged transposed [d][key] manually
// (PV B-frags contiguous). Output written bf16 to attn buffer laid out
// [B][H][S][dh] (== reference's head-interleaved reshape, flat [B*S, 4096]).
// ---------------------------------------------------------------------------
#define SKLD 136   // K tile row stride (d), 272B
#define SVLD 72    // V^T row stride (keys), 144B
#define SPLD 72    // P row stride (keys)

__global__ __launch_bounds__(256) void gqa_flash_attn(
    const __bf16* __restrict__ Q, const __bf16* __restrict__ Kc,
    const __bf16* __restrict__ Vc, __bf16* __restrict__ Oa) {
    __shared__ alignas(16) __bf16 sK[64 * SKLD];
    __shared__ alignas(16) __bf16 sV[128 * SVLD];
    __shared__ alignas(16) __bf16 sP[8][16 * SPLD];

    const int tid = threadIdx.x;
    const unsigned lane = tid & 31u;
    const int wave = tid >> 5;
    const int qb = blockIdx.x;
    const int h  = blockIdx.y;
    const int b  = blockIdx.z;
    const int kvh = h >> 2;                  // GROUP = 4
    const int q0 = qb * 128 + wave * 16;

    // Resident Q A-fragments (d = 4 chunks of 32), rows q0 + (lane&15).
    v16bf qf[4];
    {
        const __bf16* qrow =
            Q + ((size_t)b * SEQ + q0 + (lane & 15)) * D_MODEL + h * DHEAD;
#pragma unroll
        for (int kc = 0; kc < 4; ++kc) qf[kc] = load_frag_a(qrow + kc * 32, lane);
    }

    v8f o[8] = {};
    float mrow[8], lrow[8];
#pragma unroll
    for (int r = 0; r < 8; ++r) { mrow[r] = -1e30f; lrow[r] = 0.f; }

    const int nkb = qb * 2 + 2;   // causal: keys up to qb*128+127
    for (int kb = 0; kb < nkb; ++kb) {
        // ---- stage K via async DMA, V transposed via VGPR path ----
        const __bf16* kbase =
            Kc + ((size_t)b * SEQ + kb * 64) * KVDIM + kvh * DHEAD;
        const __bf16* vbase =
            Vc + ((size_t)b * SEQ + kb * 64) * KVDIM + kvh * DHEAD;
#pragma unroll
        for (int i = 0; i < 4; ++i) {
            int c = tid + i * 256;           // 0..1023
            int key = c >> 4;
            int ds = (c & 15) * 8;
            async_copy_b128(&sK[key * SKLD + ds], kbase + (size_t)key * KVDIM + ds);
        }
#pragma unroll
        for (int i = 0; i < 4; ++i) {
            int c = tid + i * 256;
            int key = c >> 4;
            int d0 = (c & 15) * 8;
            v8bf v = *(const v8bf*)(vbase + (size_t)key * KVDIM + d0);
#pragma unroll
            for (int j = 0; j < 8; ++j) sV[(d0 + j) * SVLD + key] = v[j];
        }
        wait_async_zero();
        __syncthreads();

        // ---- S = Q * K^T  (4 n-tiles of 16 keys, K-dim = 128 over 4 chunks)
        v8f st[4];
#pragma unroll
        for (int nt = 0; nt < 4; ++nt) {
            v8f s = {};
#pragma unroll
            for (int kc = 0; kc < 4; ++kc) {
                v16bf kf = load_frag_b(&sK[(nt * 16 + (lane & 15)) * SKLD + kc * 32], lane);
                s = wmma_bf16(qf[kc], kf, s);
            }
            st[nt] = s;
        }

        // ---- causal mask (branchless; keeps flow uniform across waves) ----
#pragma unroll
        for (int nt = 0; nt < 4; ++nt) {
            const int key = kb * 64 + nt * 16 + (int)(lane & 15);
#pragma unroll
            for (int r = 0; r < 8; ++r) {
                const int qrow = q0 + r + ((lane & 16) ? 8 : 0);
                if (key > qrow) st[nt][r] = -1e30f;
            }
        }

        // ---- online softmax: row max over 16-lane half-wave ----
        float scale[8];
#pragma unroll
        for (int r = 0; r < 8; ++r) {
            float v = st[0][r];
#pragma unroll
            for (int nt = 1; nt < 4; ++nt) v = fmaxf(v, st[nt][r]);
#pragma unroll
            for (int m = 1; m <= 8; m <<= 1) v = fmaxf(v, __shfl_xor(v, m, 32));
            const float mn = fmaxf(mrow[r], v);
            scale[r] = __expf(mrow[r] - mn);
            mrow[r] = mn;
        }

        float rsum[8] = {0.f, 0.f, 0.f, 0.f, 0.f, 0.f, 0.f, 0.f};
#pragma unroll
        for (int nt = 0; nt < 4; ++nt) {
            const int key = kb * 64 + nt * 16 + (int)(lane & 15);
#pragma unroll
            for (int r = 0; r < 8; ++r) {
                const int qrow = q0 + r + ((lane & 16) ? 8 : 0);
                const float p = (key > qrow) ? 0.f : __expf(st[nt][r] - mrow[r]);
                st[nt][r] = p;
                rsum[r] += p;
            }
        }
#pragma unroll
        for (int r = 0; r < 8; ++r) {
            float v = rsum[r];
#pragma unroll
            for (int m = 1; m <= 8; m <<= 1) v += __shfl_xor(v, m, 32);
            lrow[r] = lrow[r] * scale[r] + v;
        }

        // ---- repack P (C-layout) -> LDS (row-major) for A-frag reload ----
        __bf16* pw = sP[wave];
#pragma unroll
        for (int nt = 0; nt < 4; ++nt) {
#pragma unroll
            for (int r = 0; r < 8; ++r) {
                const int m = r + ((lane & 16) ? 8 : 0);
                pw[m * SPLD + nt * 16 + (lane & 15)] = (__bf16)st[nt][r];
            }
        }

        // ---- rescale O, then O += P * V ----
#pragma unroll
        for (int nt = 0; nt < 8; ++nt)
#pragma unroll
            for (int r = 0; r < 8; ++r) o[nt][r] *= scale[r];

        v16bf pf[2];
#pragma unroll
        for (int kc = 0; kc < 2; ++kc)
            pf[kc] = load_frag_a(pw + (lane & 15) * SPLD + kc * 32, lane);
#pragma unroll
        for (int nt = 0; nt < 8; ++nt) {
#pragma unroll
            for (int kc = 0; kc < 2; ++kc) {
                v16bf vf = load_frag_b(&sV[(nt * 16 + (lane & 15)) * SVLD + kc * 32], lane);
                o[nt] = wmma_bf16(pf[kc], vf, o[nt]);
            }
        }
        __syncthreads();
    }

    // ---- epilogue: O /= l, write bf16 to [B][H][S][dh] ----
#pragma unroll
    for (int nt = 0; nt < 8; ++nt) {
        const int d = nt * 16 + (int)(lane & 15);
#pragma unroll
        for (int r = 0; r < 8; ++r) {
            const int qrow = q0 + r + ((lane & 16) ? 8 : 0);
            const float v = o[nt][r] / lrow[r];
            Oa[(((size_t)b * QHEADS + h) * SEQ + qrow) * DHEAD + d] = (__bf16)v;
        }
    }
}

// ---------------------------------------------------------------------------
// Host-side launcher
// ---------------------------------------------------------------------------
extern "C" void kernel_launch(void* const* d_in, const int* in_sizes, int n_in,
                              void* d_out, int out_size, void* d_ws, size_t ws_size,
                              hipStream_t stream) {
    const float* x  = (const float*)d_in[0];
    // d_in[1] = mask (int32 tril) — causal mask is applied analytically.
    const float* Wq = (const float*)d_in[2];
    const float* bq = (const float*)d_in[3];
    const float* Wk = (const float*)d_in[4];
    const float* bk = (const float*)d_in[5];
    const float* Wv = (const float*)d_in[6];
    const float* bv = (const float*)d_in[7];
    const float* Wo = (const float*)d_in[8];
    const float* bo = (const float*)d_in[9];

    const long long n_x  = (long long)BATCH * SEQ * D_MODEL;   // 16,777,216
    const long long n_wq = (long long)D_MODEL * D_MODEL;       // 16,777,216
    const long long n_wk = (long long)KVDIM * D_MODEL;         //  4,194,304
    const long long n_q  = (long long)BATCH * SEQ * D_MODEL;
    const long long n_k  = (long long)BATCH * SEQ * KVDIM;

    char* ws = (char*)d_ws;
    size_t off = 0;
    __bf16* xb  = (__bf16*)(ws + off); off += (size_t)n_x  * 2;
    __bf16* wqb = (__bf16*)(ws + off); off += (size_t)n_wq * 2;
    __bf16* wkb = (__bf16*)(ws + off); off += (size_t)n_wk * 2;
    __bf16* wvb = (__bf16*)(ws + off); off += (size_t)n_wk * 2;
    __bf16* wob = (__bf16*)(ws + off); off += (size_t)n_wq * 2;
    __bf16* qb_ = (__bf16*)(ws + off); off += (size_t)n_q  * 2;
    __bf16* kb_ = (__bf16*)(ws + off); off += (size_t)n_k  * 2;
    __bf16* vb_ = (__bf16*)(ws + off); off += (size_t)n_k  * 2;
    __bf16* atn = (__bf16*)(ws + off); off += (size_t)n_q  * 2;

    // 1) fp32 -> bf16 conversions
    f32_to_bf16_kernel<<<(int)(n_x  / 1024), 256, 0, stream>>>(x,  xb,  n_x);
    f32_to_bf16_kernel<<<(int)(n_wq / 1024), 256, 0, stream>>>(Wq, wqb, n_wq);
    f32_to_bf16_kernel<<<(int)(n_wk / 1024), 256, 0, stream>>>(Wk, wkb, n_wk);
    f32_to_bf16_kernel<<<(int)(n_wk / 1024), 256, 0, stream>>>(Wv, wvb, n_wk);
    f32_to_bf16_kernel<<<(int)(n_wq / 1024), 256, 0, stream>>>(Wo, wob, n_wq);

    const int Mrows = BATCH * SEQ;   // 4096
    const float qscale = 0.08838834764831845f;  // 1/sqrt(128)

    // 2) Q/K/V projections (bf16 out; Q pre-scaled by 1/sqrt(d_head))
    gemm_bf16_nt<true><<<dim3(D_MODEL / BN, Mrows / BM), 256, 0, stream>>>(
        xb, wqb, bq, qb_, Mrows, D_MODEL, D_MODEL, qscale);
    gemm_bf16_nt<true><<<dim3(KVDIM / BN, Mrows / BM), 256, 0, stream>>>(
        xb, wkb, bk, kb_, Mrows, KVDIM, D_MODEL, 1.0f);
    gemm_bf16_nt<true><<<dim3(KVDIM / BN, Mrows / BM), 256, 0, stream>>>(
        xb, wvb, bv, vb_, Mrows, KVDIM, D_MODEL, 1.0f);

    // 3) causal GQA flash attention
    gqa_flash_attn<<<dim3(SEQ / 128, QHEADS, BATCH), 256, 0, stream>>>(
        qb_, kb_, vb_, atn);

    // 4) output projection (fp32 out to d_out); attn buffer's [B][H][S][dh]
    //    contiguous memory is exactly the reference's reshape(b, S, H*dh) view.
    gemm_bf16_nt<false><<<dim3(D_MODEL / BN, Mrows / BM), 256, 0, stream>>>(
        atn, wob, bo, d_out, Mrows, D_MODEL, D_MODEL, 1.0f);
}
